// RNNSynthesis_85023172592083
// MI455X (gfx1250) — compile-verified, matrix operand order checked by reading, not run on hardware
//
#include <hip/hip_runtime.h>
#include <hip/hip_bf16.h>

typedef __bf16 bf16_t;
typedef __attribute__((ext_vector_type(16))) __bf16 v16bf;
typedef __attribute__((ext_vector_type(8)))  float  v8f;

#define NWG 25
#define TB  256

constexpr int Bn = 64, Tn = 1024, Hn = 400, Vn = 77, On = 121;
constexpr int ROW2 = 512;   // in2s/in3s row: [x(0..2) pad h(16..415) win(416..492) pad]
constexpr int ROW3 = 416;   // out3s row: [h(0..399) pad]
constexpr int KP1  = 480;   // lstm1 K: x(0..2) win(3..79) h(80..479)
constexpr int KP23 = 960;   // lstm2/3 K: stream row (0..511) + h_prev(512..911) + pad

union FragU { v16bf v; uint4 q[2]; };

__device__ inline v16bf ldfrag(const bf16_t* p, int kb) {
  FragU f;
  f.q[0] = *reinterpret_cast<const uint4*>(p + kb);
  f.q[1] = *reinterpret_cast<const uint4*>(p + 16 + kb);
  return f.v;
}

__device__ inline v8f wmma_bf16(v16bf a, v16bf b, v8f c) {
  return __builtin_amdgcn_wmma_f32_16x16x32_bf16(false, a, false, b, (short)0, c, false, false);
}

__device__ inline float sigm(float x) { return 1.f / (1.f + __expf(-x)); }

__device__ inline void grid_barrier(int* cnt, int target) {
  __threadfence();
  __syncthreads();
  if (threadIdx.x == 0) {
    __hip_atomic_fetch_add(cnt, 1, __ATOMIC_RELEASE, __HIP_MEMORY_SCOPE_AGENT);
    while (__hip_atomic_load(cnt, __ATOMIC_ACQUIRE, __HIP_MEMORY_SCOPE_AGENT) < target)
      __builtin_amdgcn_s_sleep(2);
  }
  __syncthreads();
  __threadfence();
}

// ---------------- prep kernels ----------------
__global__ void zero_cnt(int* cnt) { if (threadIdx.x < 8) cnt[threadIdx.x] = 0; }

__global__ void prep_wcat1(const float* __restrict__ W1, const float* __restrict__ U1,
                           bf16_t* __restrict__ out) {
  int total = 1600 * KP1;
  for (int idx = blockIdx.x * blockDim.x + threadIdx.x; idx < total; idx += gridDim.x * blockDim.x) {
    int n = idx / KP1, k = idx % KP1;
    float v = (k < 80) ? W1[k * 1600 + n] : U1[(k - 80) * 1600 + n];
    out[idx] = (bf16_t)v;
  }
}

__global__ void prep_wcat23(const float* __restrict__ W, const float* __restrict__ U,
                            bf16_t* __restrict__ out) {
  int total = 1600 * KP23;
  for (int idx = blockIdx.x * blockDim.x + threadIdx.x; idx < total; idx += gridDim.x * blockDim.x) {
    int n = idx / KP23, k = idx % KP23;
    float v = 0.f;
    if (k < 3)                  v = W[k * 1600 + n];                 // x
    else if (k >= 16 && k < 416) v = W[(3 + k - 16) * 1600 + n];     // prev-layer h
    else if (k >= 416 && k < 493) v = W[(403 + k - 416) * 1600 + n]; // window
    else if (k >= 512 && k < 912) v = U[(k - 512) * 1600 + n];       // recurrent h
    out[idx] = (bf16_t)v;
  }
}

__global__ void prep_ww(const float* __restrict__ Ww, float* __restrict__ out) {
  int total = 30 * Hn;
  for (int idx = blockIdx.x * blockDim.x + threadIdx.x; idx < total; idx += gridDim.x * blockDim.x) {
    int j = idx / Hn, k = idx % Hn;
    out[idx] = Ww[k * 30 + j];
  }
}

__global__ void prep_wo(const float* __restrict__ Wo, bf16_t* __restrict__ out) {
  int total = 3 * 128 * ROW3;
  for (int idx = blockIdx.x * blockDim.x + threadIdx.x; idx < total; idx += gridDim.x * blockDim.x) {
    int p = idx / (128 * ROW3);
    int r = idx % (128 * ROW3);
    int n = r / ROW3, k = r % ROW3;
    float v = (n < On && k < Hn) ? Wo[(p * Hn + k) * On + n] : 0.f;
    out[idx] = (bf16_t)v;
  }
}

__global__ void prep_x(const float* __restrict__ inputs, bf16_t* __restrict__ in2s,
                       bf16_t* __restrict__ in3s) {
  int total = Tn * Bn * 3;
  for (int idx = blockIdx.x * blockDim.x + threadIdx.x; idx < total; idx += gridDim.x * blockDim.x) {
    int t = idx / (Bn * 3);
    int r = idx % (Bn * 3);
    int b = r / 3, c = r % 3;
    bf16_t v = (bf16_t)inputs[b * (Tn * 3) + t * 3 + c];
    in2s[(size_t)t * Bn * ROW2 + b * ROW2 + c] = v;
    in3s[(size_t)t * Bn * ROW2 + b * ROW2 + c] = v;
  }
}

// ---------------- LSTM1 + attention (persistent, 25 WGs, LDS-resident weights) ----------------
__global__ void lstm1_kernel(bf16_t* __restrict__ in2s, bf16_t* __restrict__ in3s,
                             const bf16_t* __restrict__ Wt, const float* __restrict__ bias,
                             const float* __restrict__ Wwt, const float* __restrict__ bw,
                             const int* __restrict__ text, const float* __restrict__ tmask,
                             int* cnt) {
  extern __shared__ char smem[];
  bf16_t* X    = (bf16_t*)smem;                           // [64][480]
  bf16_t* Wl   = (bf16_t*)(smem + 64 * KP1 * 2);          // [64][480] weight slice
  float*  Z    = (float*)(smem + 2 * 64 * KP1 * 2);       // [64][64]
  float*  cst  = Z + 64 * 64;                             // [64][16]
  float*  winF = cst + 64 * 16;                           // [64][77]
  float*  mixb = winF + 64 * 77;                          // [64][30]
  float*  kap  = mixb + 64 * 30;                          // [64][10]
  int*    txL  = (int*)(kap + 64 * 10);                   // [64][64]
  float*  mkL  = (float*)(txL + 64 * 64);                 // [64][64]

  const int wg = blockIdx.x, tid = threadIdx.x;
  const int lane = tid & 31, wv = tid >> 5;
  const int lr = lane & 15;
  const int kb = (lane & 16) ? 8 : 0;

  for (int i = tid; i < 64 * KP1 / 2; i += TB) ((unsigned int*)X)[i] = 0u;
  for (int i = tid; i < 64 * 16; i += TB) cst[i] = 0.f;
  for (int i = tid; i < 64 * 10; i += TB) kap[i] = 0.f;
  for (int i = tid; i < 64 * 64; i += TB) { txL[i] = text[i]; mkL[i] = tmask[i]; }
  // stage this WG's weight slice (4 gates x 16 units) into LDS once
  for (int i = tid; i < 64 * (KP1 / 8); i += TB) {
    int lrow = i / (KP1 / 8), cc = (i % (KP1 / 8)) * 8;
    int g = lrow >> 4, j = lrow & 15;
    size_t grow = (size_t)(g * Hn + wg * 16 + j);
    *(uint4*)&Wl[lrow * KP1 + cc] = *(const uint4*)&Wt[grow * KP1 + cc];
  }
  __syncthreads();
  for (int i = tid; i < 64 * 3; i += TB) {     // x_0
    int b = i / 3, c = i % 3;
    X[b * KP1 + c] = in2s[b * ROW2 + c];
  }
  __syncthreads();

  for (int t = 0; t < Tn; ++t) {
    asm volatile("" ::: "memory");   // keep LDS frag loads inside the time loop
    size_t t2 = (size_t)t * Bn * ROW2;
    // z-slice GEMM: wave handles one M-tile, two gate tiles (A reused)
    {
      const int m  = wv & 3;
      const int g0 = (wv >> 2) * 2;
      const bf16_t* xrow = X + (m * 16 + lr) * KP1;
      const bf16_t* w0 = Wl + ((g0 * 16)      + lr) * KP1;
      const bf16_t* w1 = Wl + (((g0 + 1) * 16) + lr) * KP1;
      v8f acc0 = {}, acc1 = {};
      for (int kt = 0; kt < KP1 / 32; ++kt) {
        v16bf a = ldfrag(xrow + kt * 32, kb);
        acc0 = wmma_bf16(a, ldfrag(w0 + kt * 32, kb), acc0);
        acc1 = wmma_bf16(a, ldfrag(w1 + kt * 32, kb), acc1);
      }
      int rb = m * 16 + ((lane & 16) ? 8 : 0);
      int zc0 = g0 * 16 + lr;
#pragma unroll
      for (int v = 0; v < 8; ++v) {
        Z[(rb + v) * 64 + zc0]      = acc0[v];
        Z[(rb + v) * 64 + zc0 + 16] = acc1[v];
      }
    }
    __syncthreads();
    // gates -> h slice -> in2s[t]
    for (int it = tid; it < 64 * 16; it += TB) {
      int b = it >> 4, u = it & 15;
      int hu = wg * 16 + u;
      float zi = Z[b * 64 + u]       + bias[hu];
      float zf = Z[b * 64 + 16 + u]  + bias[Hn + hu];
      float zg = Z[b * 64 + 32 + u]  + bias[2 * Hn + hu];
      float zo = Z[b * 64 + 48 + u]  + bias[3 * Hn + hu];
      float c = sigm(zf) * cst[it] + sigm(zi) * tanhf(zg);
      cst[it] = c;
      float h = sigm(zo) * tanhf(c);
      in2s[t2 + b * ROW2 + 16 + hu] = (bf16_t)h;
    }
    grid_barrier(cnt, NWG * (t + 1));
    // reload full h(t) into X[:,80:480] (h_prev for t+1, also attention input)
    for (int i = tid; i < 64 * 50; i += TB) {
      int b = i / 50, d = (i % 50) * 8;
      *(uint4*)&X[b * KP1 + 80 + d] = *(const uint4*)&in2s[t2 + b * ROW2 + 16 + d];
    }
    __syncthreads();
    // mixture params: exp(h @ Ww + bw)
    for (int i = tid; i < 64 * 30; i += TB) {
      int b = i / 30, j = i % 30;
      float s = bw[j];
      const bf16_t* hp = X + b * KP1 + 80;
      const float* wp = Wwt + j * Hn;
      for (int k = 0; k < Hn; ++k) s += (float)hp[k] * wp[k];
      mixb[i] = __expf(s);
    }
    __syncthreads();
    for (int i = tid; i < 64 * 10; i += TB) kap[i] += mixb[(i / 10) * 30 + 20 + (i % 10)];
    for (int i = tid; i < 64 * 77; i += TB) winF[i] = 0.f;
    __syncthreads();
    // phi + one-hot scatter into window
    for (int i = tid; i < 64 * 64; i += TB) {
      int b = i >> 6, u = i & 63;
      float phi = 0.f;
#pragma unroll
      for (int k = 0; k < 10; ++k) {
        float al = mixb[b * 30 + k];
        float be = mixb[b * 30 + 10 + k];
        float d  = kap[b * 10 + k] - (float)u;
        phi += al * __expf(-be * d * d);
      }
      phi *= mkL[i];
      atomicAdd(&winF[b * Vn + txL[i]], phi);
    }
    __syncthreads();
    // window -> X (next step) and streams (WG0 writes globally)
    for (int i = tid; i < 64 * Vn; i += TB) {
      int b = i / Vn, v = i % Vn;
      bf16_t w16 = (bf16_t)winF[i];
      X[b * KP1 + 3 + v] = w16;
      if (wg == 0) {
        in2s[t2 + b * ROW2 + 416 + v] = w16;
        in3s[t2 + b * ROW2 + 416 + v] = w16;
      }
    }
    if (t + 1 < Tn) {
      for (int i = tid; i < 64 * 3; i += TB) {
        int b = i / 3, c = i % 3;
        X[b * KP1 + c] = in2s[t2 + (size_t)Bn * ROW2 + b * ROW2 + c];
      }
    }
    __syncthreads();
  }
}

// ---------------- LSTM2/LSTM3 (persistent, 25 WGs, LDS-resident weights) ----------------
__global__ void lstm23_kernel(const bf16_t* __restrict__ ins, bf16_t* __restrict__ outs,
                              const bf16_t* __restrict__ Wt, const float* __restrict__ bias,
                              int* cnt, int orow, int ooff) {
  extern __shared__ char smem[];
  bf16_t* X   = (bf16_t*)smem;                        // [64][960]
  bf16_t* Wl  = (bf16_t*)(smem + 64 * KP23 * 2);      // [64][960] weight slice
  float*  Z   = (float*)(smem + 2 * 64 * KP23 * 2);   // [64][64]
  float*  cst = Z + 64 * 64;                          // [64][16]

  const int wg = blockIdx.x, tid = threadIdx.x;
  const int lane = tid & 31, wv = tid >> 5;
  const int lr = lane & 15;
  const int kb = (lane & 16) ? 8 : 0;

  for (int i = tid; i < 64 * KP23 / 2; i += TB) ((unsigned int*)X)[i] = 0u;
  for (int i = tid; i < 64 * 16; i += TB) cst[i] = 0.f;
  for (int i = tid; i < 64 * (KP23 / 8); i += TB) {
    int lrow = i / (KP23 / 8), cc = (i % (KP23 / 8)) * 8;
    int g = lrow >> 4, j = lrow & 15;
    size_t grow = (size_t)(g * Hn + wg * 16 + j);
    *(uint4*)&Wl[lrow * KP23 + cc] = *(const uint4*)&Wt[grow * KP23 + cc];
  }
  __syncthreads();

  for (int t = 0; t < Tn; ++t) {
    asm volatile("" ::: "memory");   // keep LDS frag loads inside the time loop
    size_t tb = (size_t)t * Bn * ROW2;
    for (int i = tid; i < 64 * 64; i += TB) {               // X[:,0:512] <- ins[t]
      int b = i >> 6, d = (i & 63) * 8;
      *(uint4*)&X[b * KP23 + d] = *(const uint4*)&ins[tb + b * ROW2 + d];
    }
    if (t > 0) {                                            // X[:,512:912] <- own h(t-1)
      size_t hb = (size_t)(t - 1) * Bn * orow;
      for (int i = tid; i < 64 * 50; i += TB) {
        int b = i / 50, d = (i % 50) * 8;
        *(uint4*)&X[b * KP23 + 512 + d] = *(const uint4*)&outs[hb + b * orow + ooff + d];
      }
    }
    __syncthreads();
    {
      const int m  = wv & 3;
      const int g0 = (wv >> 2) * 2;
      const bf16_t* xrow = X + (m * 16 + lr) * KP23;
      const bf16_t* w0 = Wl + ((g0 * 16)       + lr) * KP23;
      const bf16_t* w1 = Wl + (((g0 + 1) * 16) + lr) * KP23;
      v8f acc0 = {}, acc1 = {};
      for (int kt = 0; kt < KP23 / 32; ++kt) {
        v16bf a = ldfrag(xrow + kt * 32, kb);
        acc0 = wmma_bf16(a, ldfrag(w0 + kt * 32, kb), acc0);
        acc1 = wmma_bf16(a, ldfrag(w1 + kt * 32, kb), acc1);
      }
      int rb = m * 16 + ((lane & 16) ? 8 : 0);
      int zc0 = g0 * 16 + lr;
#pragma unroll
      for (int v = 0; v < 8; ++v) {
        Z[(rb + v) * 64 + zc0]      = acc0[v];
        Z[(rb + v) * 64 + zc0 + 16] = acc1[v];
      }
    }
    __syncthreads();
    for (int it = tid; it < 64 * 16; it += TB) {
      int b = it >> 4, u = it & 15;
      int hu = wg * 16 + u;
      float zi = Z[b * 64 + u]       + bias[hu];
      float zf = Z[b * 64 + 16 + u]  + bias[Hn + hu];
      float zg = Z[b * 64 + 32 + u]  + bias[2 * Hn + hu];
      float zo = Z[b * 64 + 48 + u]  + bias[3 * Hn + hu];
      float c = sigm(zf) * cst[it] + sigm(zi) * tanhf(zg);
      cst[it] = c;
      float h = sigm(zo) * tanhf(c);
      outs[(size_t)t * Bn * orow + b * orow + ooff + hu] = (bf16_t)h;
    }
    grid_barrier(cnt, NWG * (t + 1));
  }
}

// ---------------- output projection: y = out1@Wo0 + out2@Wo1 + out3@Wo2 + bo ----------------
__global__ void out_gemm(const bf16_t* __restrict__ in2s, const bf16_t* __restrict__ in3s,
                         const bf16_t* __restrict__ out3s, const bf16_t* __restrict__ Wotp,
                         const float* __restrict__ bo, float* __restrict__ out) {
  __shared__ bf16_t A3[16 * 1248];      // 16 rows x [seg0(416) seg1(416) seg2(416)], k>=400 zero
  const int mt = blockIdx.x;            // 4096 row tiles of 16
  const int tid = threadIdx.x;
  const int lane = tid & 31, w = tid >> 5;
  const int lr = lane & 15;
  const int kb = (lane & 16) ? 8 : 0;
  const int n0 = w * 16;

  // stage A panel into LDS (zero pads past k=400)
  for (int i = tid; i < 16 * 156; i += 256) {
    int r = i / 156, cc = i % 156;
    int seg = cc / 52;
    int k8 = (cc % 52) * 8;
    uint4 val = make_uint4(0u, 0u, 0u, 0u);
    if (k8 < Hn) {
      if (seg == 0)      val = *(const uint4*)&in2s[(size_t)(mt * 16 + r) * ROW2 + 16 + k8];
      else if (seg == 1) val = *(const uint4*)&in3s[(size_t)(mt * 16 + r) * ROW2 + 16 + k8];
      else               val = *(const uint4*)&out3s[(size_t)(mt * 16 + r) * ROW3 + k8];
    }
    *(uint4*)&A3[r * 1248 + seg * 416 + k8] = val;
  }
  __syncthreads();

  v8f acc = {};
#pragma unroll
  for (int p = 0; p < 3; ++p) {
    const bf16_t* arow = A3 + lr * 1248 + p * 416;
    const bf16_t* wrow = Wotp + ((size_t)p * 128 + n0 + lr) * ROW3;
    for (int kt = 0; kt < 13; ++kt) {
      v16bf a  = ldfrag(arow + kt * 32, kb);
      v16bf bm = ldfrag(wrow + kt * 32, kb);
      acc = wmma_bf16(a, bm, acc);
    }
  }
  int n = n0 + lr;
  if (n < On) {
    int rb = mt * 16 + ((lane & 16) ? 8 : 0);
    float bias = bo[n];
#pragma unroll
    for (int v = 0; v < 8; ++v) {
      int r = rb + v;
      int t = r >> 6, b = r & 63;
      out[(size_t)b * Tn * On + (size_t)t * On + n] = acc[v] + bias;
    }
  }
}

extern "C" void kernel_launch(void* const* d_in, const int* in_sizes, int n_in,
                              void* d_out, int out_size, void* d_ws, size_t ws_size,
                              hipStream_t stream) {
  (void)in_sizes; (void)n_in; (void)out_size; (void)ws_size;
  const float* inputs = (const float*)d_in[0];
  const int*   text   = (const int*)d_in[1];
  const float* tmask  = (const float*)d_in[2];
  const float* W1 = (const float*)d_in[3];
  const float* U1 = (const float*)d_in[4];
  const float* b1 = (const float*)d_in[5];
  const float* Ww = (const float*)d_in[6];
  const float* bw = (const float*)d_in[7];
  const float* W2 = (const float*)d_in[8];
  const float* U2 = (const float*)d_in[9];
  const float* b2 = (const float*)d_in[10];
  const float* W3 = (const float*)d_in[11];
  const float* U3 = (const float*)d_in[12];
  const float* b3 = (const float*)d_in[13];
  const float* Wo = (const float*)d_in[14];
  const float* bo = (const float*)d_in[15];
  float* out = (float*)d_out;

  char* ws = (char*)d_ws;
  size_t off = 0;
  auto take = [&](size_t bytes) -> void* {
    void* p = (void*)(ws + off);
    off += (bytes + 255) & ~(size_t)255;
    return p;
  };
  int*    cnt   = (int*)take(256);
  bf16_t* Wcat1 = (bf16_t*)take((size_t)1600 * KP1 * 2);
  bf16_t* Wcat2 = (bf16_t*)take((size_t)1600 * KP23 * 2);
  bf16_t* Wcat3 = (bf16_t*)take((size_t)1600 * KP23 * 2);
  float*  Wwt   = (float*)take((size_t)30 * Hn * 4);
  bf16_t* Wotp  = (bf16_t*)take((size_t)3 * 128 * ROW3 * 2);
  bf16_t* in2s  = (bf16_t*)take((size_t)Tn * Bn * ROW2 * 2);
  bf16_t* in3s  = (bf16_t*)take((size_t)Tn * Bn * ROW2 * 2);
  bf16_t* out3s = (bf16_t*)take((size_t)Tn * Bn * ROW3 * 2);

  zero_cnt<<<1, 64, 0, stream>>>(cnt);
  prep_wcat1<<<512, 256, 0, stream>>>(W1, U1, Wcat1);
  prep_wcat23<<<960, 256, 0, stream>>>(W2, U2, Wcat2);
  prep_wcat23<<<960, 256, 0, stream>>>(W3, U3, Wcat3);
  prep_ww<<<47, 256, 0, stream>>>(Ww, Wwt);
  prep_wo<<<624, 256, 0, stream>>>(Wo, Wotp);
  prep_x<<<768, 256, 0, stream>>>(inputs, in2s, in3s);

  size_t sm1  = (size_t)2 * 64 * KP1 * 2 + 64 * 64 * 4 + 64 * 16 * 4 + 64 * 77 * 4 +
                64 * 30 * 4 + 64 * 10 * 4 + 64 * 64 * 4 + 64 * 64 * 4;   // 206,080 B
  size_t sm23 = (size_t)2 * 64 * KP23 * 2 + 64 * 64 * 4 + 64 * 16 * 4;   // 266,240 B

  lstm1_kernel<<<NWG, TB, sm1, stream>>>(in2s, in3s, Wcat1, b1, Wwt, bw, text, tmask, cnt);
  lstm23_kernel<<<NWG, TB, sm23, stream>>>(in2s, in3s, Wcat2, b2, cnt + 1, ROW2, 16);
  lstm23_kernel<<<NWG, TB, sm23, stream>>>(in3s, out3s, Wcat3, b3, cnt + 2, ROW3, 0);
  out_gemm<<<4096, 256, 0, stream>>>(in2s, in3s, out3s, Wotp, bo, out);
}
